// GeoGNNBlock_42159398977846
// MI455X (gfx1250) — matrix operand order repaired and checked by reading, not compiled, not run on hardware
//
#include <hip/hip_runtime.h>
#include <hip/hip_bf16.h>

#define NNODES 40000
#define NEDGES 640000
#define DIM    128
#define NGRAPH 512

typedef __attribute__((ext_vector_type(16))) __bf16 v16bf;
typedef __attribute__((ext_vector_type(8)))  __bf16 v8bf;
typedef __attribute__((ext_vector_type(8)))  float  v8f;

// ---------------- zero: edge counts + graph degree ----------------
__global__ void k_zero(int* __restrict__ cnt, float* __restrict__ deg) {
    int i = blockIdx.x * blockDim.x + threadIdx.x;
    if (i < NNODES) cnt[i] = 0;
    if (i < NGRAPH) deg[i] = 0.0f;
}

// ---------------- CSR build step 1: histogram of dst ----------------
__global__ void k_count(const long long* __restrict__ ei, int* __restrict__ cnt) {
    int e = blockIdx.x * blockDim.x + threadIdx.x;
    if (e < NEDGES) atomicAdd(&cnt[(int)ei[NEDGES + e]], 1);
}

// ---------------- CSR build step 2: single-block exclusive scan ----------------
__global__ __launch_bounds__(1024) void k_scan(const int* __restrict__ cnt,
                                               int* __restrict__ rowstart,
                                               int* __restrict__ cursor) {
    __shared__ int buf[1024];
    __shared__ int carry;
    const int tid = threadIdx.x;
    if (tid == 0) carry = 0;
    __syncthreads();
    for (int base = 0; base < NNODES; base += 1024) {
        int i = base + tid;
        int v = (i < NNODES) ? cnt[i] : 0;
        buf[tid] = v;
        __syncthreads();
        #pragma unroll
        for (int off = 1; off < 1024; off <<= 1) {
            int t = (tid >= off) ? buf[tid - off] : 0;
            __syncthreads();
            buf[tid] += t;
            __syncthreads();
        }
        int incl = buf[tid];
        int excl = incl - v + carry;          // reads carry before update below
        if (i < NNODES) { rowstart[i] = excl; cursor[i] = excl; }
        __syncthreads();
        if (tid == 1023) carry += incl;       // chunk total
        __syncthreads();
    }
    if (tid == 0) rowstart[NNODES] = carry;   // == NEDGES
}

// ---------------- CSR build step 3: scatter edge ids into rows ----------------
__global__ void k_fill(const long long* __restrict__ ei, int* __restrict__ cursor,
                       int* __restrict__ elist) {
    int e = blockIdx.x * blockDim.x + threadIdx.x;
    if (e < NEDGES) {
        int d = (int)ei[NEDGES + e];
        int p = atomicAdd(&cursor[d], 1);
        elist[p] = e;
    }
}

// ---------------- atomic-free aggregation: aggr[n] = sum over in-edges ----------------
// One wave per node: 32 lanes x float4 = 128 features. Each edge costs one
// coalesced 512B edge_attr row (HBM stream) + one x row (L2-resident gather).
__global__ void k_gather(const float* __restrict__ x, const long long* __restrict__ ei,
                         const float* __restrict__ ea, const int* __restrict__ rowstart,
                         const int* __restrict__ elist, float* __restrict__ aggr) {
    int node = blockIdx.x * (blockDim.x >> 5) + (threadIdx.x >> 5);
    if (node >= NNODES) return;
    int lane = threadIdx.x & 31;
    int c = lane * 4;
    float4 acc; acc.x = acc.y = acc.z = acc.w = 0.0f;
    int p0 = rowstart[node], p1 = rowstart[node + 1];
    for (int p = p0; p < p1; ++p) {
        int e = elist[p];
        int s = (int)ei[e];
        float4 xv = *(const float4*)(x  + (size_t)s * DIM + c);
        float4 av = *(const float4*)(ea + (size_t)e * DIM + c);
        acc.x += xv.x + av.x;
        acc.y += xv.y + av.y;
        acc.z += xv.z + av.z;
        acc.w += xv.w + av.w;
    }
    *(float4*)(aggr + (size_t)node * DIM + c) = acc;
}

// ---------------- graph degree + per-node GraphNorm scale ----------------
__global__ void k_deg(const long long* __restrict__ batch, float* __restrict__ deg) {
    int i = blockIdx.x * blockDim.x + threadIdx.x;
    if (i < NNODES) atomicAdd(&deg[(int)batch[i]], 1.0f);
}
__global__ void k_scale(const long long* __restrict__ batch, const float* __restrict__ deg,
                        float* __restrict__ scl) {
    int i = blockIdx.x * blockDim.x + threadIdx.x;
    if (i < NNODES) scl[i] = rsqrtf(fmaxf(deg[(int)batch[i]], 1.0f));
}

// ---------------- pre-pack weights into WMMA B-fragment layout ----------------
// B layout (16-bit, 32x16): lane L holds col = L%16; lanes 0-15 K=0..7,16..23,
// lanes 16-31 K=8..15,24..31, 2 K-values per VGPR (sequential).
__global__ void k_pack(const float* __restrict__ w1, const float* __restrict__ w2,
                       __bf16* __restrict__ pw1, __bf16* __restrict__ pw2) {
    int t = blockIdx.x * blockDim.x + threadIdx.x;   // 4096 threads total
    int lane = t & 31, frag = t >> 5;
    int c16 = lane & 15;
    int kb  = (lane >> 4) * 8;
    if (frag < 64) {                 // w1: [128 x 256], kk 0..3, nn 0..15
        int kk = frag >> 4, nnv = frag & 15;
        int col = nnv * 16 + c16;
        __bf16* o = pw1 + (size_t)t * 16;
        #pragma unroll
        for (int i = 0; i < 16; ++i)
            o[i] = (__bf16)w1[(kk * 32 + kb + i + (i & 8)) * 256 + col];
    } else {                         // w2: [256 x 128], kk 0..7, nn 0..7
        int f = frag - 64;
        int kk = f >> 3, nnv = f & 7;
        int col = nnv * 16 + c16;
        __bf16* o = pw2 + (size_t)(f * 32 + lane) * 16;
        #pragma unroll
        for (int i = 0; i < 16; ++i)
            o[i] = (__bf16)w2[(kk * 32 + kb + i + (i & 8)) * 128 + col];
    }
}

// ---------------- fused MLP (WMMA bf16) + LN + GraphNorm + ReLU + residual ----------------
// One wave per 16-node tile; block = 4 waves; 625 blocks * 4 = 2500 tiles = 40000 nodes.
__global__ __launch_bounds__(128) void k_mlp(
    const float* __restrict__ aggr, const __bf16* __restrict__ pw1,
    const __bf16* __restrict__ pw2,
    const float* __restrict__ b1, const float* __restrict__ b2,
    const float* __restrict__ lnw, const float* __restrict__ lnb,
    const float* __restrict__ scl, const float* __restrict__ x,
    float* __restrict__ out)
{
    __shared__ __align__(16) __bf16 hbuf[4][16][264];   // 256 + 8 pad, per-wave tile
    const int lane = threadIdx.x & 31;
    const int wave = threadIdx.x >> 5;
    const int tile = blockIdx.x * 4 + wave;
    const int c16  = lane & 15;
    const int half = lane >> 4;
    const int kb   = half * 8;          // A-layout interleave base
    const int nb   = tile * 16;

    // ---- A1: 16x128 aggr tile -> bf16 A-fragments (lanes hold row = lane%16) ----
    v16bf A1[4];
    const float* ar = aggr + (size_t)(nb + c16) * DIM;
    #pragma unroll
    for (int kk = 0; kk < 4; ++kk) {
        float4 r0 = *(const float4*)(ar + kk * 32 + kb);
        float4 r1 = *(const float4*)(ar + kk * 32 + kb + 4);
        float4 r2 = *(const float4*)(ar + kk * 32 + kb + 16);
        float4 r3 = *(const float4*)(ar + kk * 32 + kb + 20);
        v16bf a;
        a[0]=(__bf16)r0.x; a[1]=(__bf16)r0.y; a[2]=(__bf16)r0.z; a[3]=(__bf16)r0.w;
        a[4]=(__bf16)r1.x; a[5]=(__bf16)r1.y; a[6]=(__bf16)r1.z; a[7]=(__bf16)r1.w;
        a[8]=(__bf16)r2.x; a[9]=(__bf16)r2.y; a[10]=(__bf16)r2.z; a[11]=(__bf16)r2.w;
        a[12]=(__bf16)r3.x; a[13]=(__bf16)r3.y; a[14]=(__bf16)r3.z; a[15]=(__bf16)r3.w;
        A1[kk] = a;
    }

    // ---- layer 1: 16x128 @ 128x256, relu, stash bf16 tile in LDS ----
    for (int nnv = 0; nnv < 16; ++nnv) {
        v8f acc = {};
        #pragma unroll
        for (int kk = 0; kk < 4; ++kk) {
            v16bf b = *(const v16bf*)(pw1 + (size_t)((kk * 16 + nnv) * 32 + lane) * 16);
            acc = __builtin_amdgcn_wmma_f32_16x16x32_bf16(
                      false, A1[kk], false, b, (short)0, acc, false, false);
        }
        float bias = b1[nnv * 16 + c16];
        #pragma unroll
        for (int v = 0; v < 8; ++v) {
            float hv = fmaxf(acc[v] + bias, 0.0f);
            hbuf[wave][v + half * 8][nnv * 16 + c16] = (__bf16)hv;   // D: col=lane%16, row=v+8*half
        }
    }
    __syncthreads();   // lane-transpose handoff through LDS

    // ---- A2: reload h from LDS in A-fragment layout ----
    v16bf A2[8];
    #pragma unroll
    for (int kk = 0; kk < 8; ++kk) {
        const __bf16* hr = &hbuf[wave][c16][kk * 32 + kb];
        v8bf lo = *(const v8bf*)hr;
        v8bf hi = *(const v8bf*)(hr + 16);
        A2[kk] = __builtin_shufflevector(lo, hi, 0,1,2,3,4,5,6,7,8,9,10,11,12,13,14,15);
    }

    // ---- layer 2: 16x256 @ 256x128 ----
    v8f acc2[8];
    #pragma unroll
    for (int nnv = 0; nnv < 8; ++nnv) {
        v8f acc = {};
        #pragma unroll
        for (int kk = 0; kk < 8; ++kk) {
            v16bf b = *(const v16bf*)(pw2 + (size_t)((kk * 8 + nnv) * 32 + lane) * 16);
            acc = __builtin_amdgcn_wmma_f32_16x16x32_bf16(
                      false, A2[kk], false, b, (short)0, acc, false, false);
        }
        acc2[nnv] = acc;
    }

    // ---- bias + LayerNorm stats (reduce across 16-lane halves with shfl_xor) ----
    float b2c[8], lw[8], lb[8];
    #pragma unroll
    for (int nnv = 0; nnv < 8; ++nnv) {
        b2c[nnv] = b2[nnv * 16 + c16];
        lw[nnv]  = lnw[nnv * 16 + c16];
        lb[nnv]  = lnb[nnv * 16 + c16];
    }
    float mu[8], rs[8];
    #pragma unroll
    for (int v = 0; v < 8; ++v) {
        float s = 0.0f, ss = 0.0f;
        #pragma unroll
        for (int nnv = 0; nnv < 8; ++nnv) {
            float val = acc2[nnv][v] + b2c[nnv];
            acc2[nnv][v] = val;
            s += val; ss += val * val;
        }
        #pragma unroll
        for (int m = 1; m <= 8; m <<= 1) {
            s  += __shfl_xor(s,  m, 32);
            ss += __shfl_xor(ss, m, 32);
        }
        float mean = s * (1.0f / DIM);
        mu[v] = mean;
        rs[v] = rsqrtf(ss * (1.0f / DIM) - mean * mean + 1e-5f);
    }

    // ---- LN + GraphNorm + ReLU + residual, single pass out ----
    #pragma unroll
    for (int v = 0; v < 8; ++v) {
        int node = nb + v + half * 8;
        float sc = scl[node];
        const float* xr = x + (size_t)node * DIM;
        float* orow = out + (size_t)node * DIM;
        #pragma unroll
        for (int nnv = 0; nnv < 8; ++nnv) {
            int col = nnv * 16 + c16;
            float val = (acc2[nnv][v] - mu[v]) * rs[v] * lw[nnv] + lb[nnv];
            val = fmaxf(val * sc, 0.0f);
            orow[col] = val + xr[col];
        }
    }
}

// ---------------- launch ----------------
extern "C" void kernel_launch(void* const* d_in, const int* in_sizes, int n_in,
                              void* d_out, int out_size, void* d_ws, size_t ws_size,
                              hipStream_t stream) {
    const float*     x     = (const float*)d_in[0];
    const long long* ei    = (const long long*)d_in[1];
    const float*     ea    = (const float*)d_in[2];
    const long long* batch = (const long long*)d_in[3];
    const float*     w1    = (const float*)d_in[4];
    const float*     b1    = (const float*)d_in[5];
    const float*     w2    = (const float*)d_in[6];
    const float*     b2    = (const float*)d_in[7];
    const float*     lnw   = (const float*)d_in[8];
    const float*     lnb   = (const float*)d_in[9];
    float* out = (float*)d_out;

    char* ws = (char*)d_ws;
    float*  aggr     = (float*)(ws);                   // 20,480,000 B
    float*  deg      = (float*)(ws + 20480000);        // 2,048 B (padded)
    float*  scl      = (float*)(ws + 20482048);        // 160,000 B
    __bf16* pw1      = (__bf16*)(ws + 20642048);       // 65,536 B (256B aligned)
    __bf16* pw2      = (__bf16*)(ws + 20707584);       // 65,536 B
    int*    cnt      = (int*)  (ws + 20773120);        // 160,000 B
    int*    rowstart = (int*)  (ws + 20933120);        // 160,004 B (+pad)
    int*    cursor   = (int*)  (ws + 21093376);        // 160,000 B
    int*    elist    = (int*)  (ws + 21253376);        // 2,560,000 B  -> end ~23.8 MB

    k_zero  <<<(NNODES + 255) / 256, 256, 0, stream>>>(cnt, deg);
    k_count <<<(NEDGES + 255) / 256, 256, 0, stream>>>(ei, cnt);
    k_scan  <<<1, 1024, 0, stream>>>(cnt, rowstart, cursor);
    k_fill  <<<(NEDGES + 255) / 256, 256, 0, stream>>>(ei, cursor, elist);
    k_gather<<<(NNODES + 7) / 8, 256, 0, stream>>>(x, ei, ea, rowstart, elist, aggr);
    k_deg   <<<(NNODES + 255) / 256, 256, 0, stream>>>(batch, deg);
    k_scale <<<(NNODES + 255) / 256, 256, 0, stream>>>(batch, deg, scl);
    k_pack  <<<16, 256, 0, stream>>>(w1, w2, pw1, pw2);
    k_mlp   <<<625, 128, 0, stream>>>(aggr, pw1, pw2, b1, b2, lnw, lnb, scl, x, out);
}
